// Attention_10866267259381
// MI455X (gfx1250) — compile-verified, matrix operand order checked by reading, not run on hardware
//
#include <hip/hip_runtime.h>

// Reference: attn = softmax(L2norm_row((q/8 @ k^T) / |global_min|)) ; out = attn @ v
// Returns (out, attn, v). The |global_min| factor cancels inside the row L2
// normalization (norm scales by the same factor), except through the eps=1e-12
// clamp which never binds for Gaussian inputs -> we compute row/||row|| directly.

#define SDIM 4096
#define DDIM 64
#define BHN  16          // B*H
#define BM   64          // rows per workgroup
#define NTHREADS 128     // 4 waves (wave32)

typedef __attribute__((ext_vector_type(16))) __bf16 v16bf;
typedef __attribute__((ext_vector_type(2)))  __bf16 v2bf;
typedef __attribute__((ext_vector_type(8)))  float  v8f;

union BFrag { v16bf v; unsigned u[8]; uint4 q[2]; };

__device__ __forceinline__ unsigned pack2bf(float lo, float hi) {
  union { v2bf v; unsigned u; } c;
  c.v.x = (__bf16)lo;          // RNE conversion; lowers to v_cvt_pk_bf16_f32
  c.v.y = (__bf16)hi;
  return c.u;
}
__device__ __forceinline__ unsigned short f2bf(float f) {
  union { __bf16 b; unsigned short u; } c;
  c.b = (__bf16)f;
  return c.u;
}
__device__ __forceinline__ void pk4(unsigned* u, float4 f, float sc) {
  u[0] = pack2bf(f.x * sc, f.y * sc);
  u[1] = pack2bf(f.z * sc, f.w * sc);
}

__global__ __launch_bounds__(NTHREADS)
void attn_fused(const float* __restrict__ q, const float* __restrict__ k,
                const float* __restrict__ v, float* __restrict__ out) {
  __shared__ unsigned short lKb[BM * DDIM];   // 8 KB: K tile, bf16, d-contiguous
  __shared__ unsigned short lVt[DDIM * 32];   // 4 KB: V tile, bf16, transposed [d][t]
  __shared__ float rowSumSq[BM];
  __shared__ float rowMax[BM];
  __shared__ float rowSum[BM];

  const int tid  = threadIdx.x;
  const int lane = tid & 31;
  const int wave = tid >> 5;      // 0..3, owns 16 rows
  const int half = lane >> 4;     // lane group for WMMA layouts
  const int ln   = lane & 15;
  const int bh   = blockIdx.y;    // head index (B*H flattened)
  const int r0   = blockIdx.x * BM;

  const size_t headQK = (size_t)bh * SDIM * DDIM;
  const float* qh = q + headQK;
  const float* kh = k + headQK;
  const float* vh = v + headQK;
  float* outH  = out + headQK;                                               // [BH,S,D]
  float* attnH = out + (size_t)BHN * SDIM * DDIM + (size_t)bh * SDIM * SDIM; // [BH,S,S]

  // ---------------- A fragments from Q rows (pre-scaled by 1/TEMPERATURE) -----
  // 16-bit A layout: lane ln row M=ln; half0 K=0..7,16..23 ; half1 K=8..15,24..31
  BFrag a0, a1;
  {
    const float* qr = qh + (size_t)(r0 + wave * 16 + ln) * DDIM + 8 * half;
    const float sc = 0.125f;
    pk4(&a0.u[0], *(const float4*)(qr + 0),  sc);
    pk4(&a0.u[2], *(const float4*)(qr + 4),  sc);
    pk4(&a0.u[4], *(const float4*)(qr + 16), sc);
    pk4(&a0.u[6], *(const float4*)(qr + 20), sc);
    pk4(&a1.u[0], *(const float4*)(qr + 32), sc);
    pk4(&a1.u[2], *(const float4*)(qr + 36), sc);
    pk4(&a1.u[4], *(const float4*)(qr + 48), sc);
    pk4(&a1.u[6], *(const float4*)(qr + 52), sc);
  }

  // ---------------- Pass A: raw scores + per-row sumsq / max ------------------
  float ssq[8], smax[8];
#pragma unroll
  for (int j = 0; j < 8; ++j) { ssq[j] = 0.f; smax[j] = -3.4e38f; }

  for (int ct = 0; ct < SDIM / 64; ++ct) {
    __syncthreads();
    { // cooperative stage of 64 K-rows (fp32 -> bf16) into LDS, ds_store_b64
      const float* src = kh + (size_t)ct * 64 * DDIM;
      unsigned* dst = (unsigned*)lKb;
#pragma unroll
      for (int i = 0; i < 8; ++i) {
        int l = i * 512 + tid * 4;                  // coalesced float4 runs
        float4 f = *(const float4*)(src + l);
        uint2 p = make_uint2(pack2bf(f.x, f.y), pack2bf(f.z, f.w));
        *(uint2*)(dst + (l >> 1)) = p;              // 8-byte aligned
      }
    }
    __syncthreads();
    const uint4* kb4 = (const uint4*)lKb;
#pragma unroll
    for (int s = 0; s < 4; ++s) {                   // four 16-column subtiles
      BFrag b0, b1;
      int base = (s * 16 + ln) * 8 + half;          // uint4 index, 16B aligned
      b0.q[0] = kb4[base + 0];                      // ds_load_b128 x4
      b0.q[1] = kb4[base + 2];
      b1.q[0] = kb4[base + 4];
      b1.q[1] = kb4[base + 6];
      v8f cs = {0.f, 0.f, 0.f, 0.f, 0.f, 0.f, 0.f, 0.f};
      cs = __builtin_amdgcn_wmma_f32_16x16x32_bf16(false, a0.v, false, b0.v,
                                                   (short)0, cs, false, false);
      cs = __builtin_amdgcn_wmma_f32_16x16x32_bf16(false, a1.v, false, b1.v,
                                                   (short)0, cs, false, false);
      int col = ct * 64 + s * 16 + ln;
#pragma unroll
      for (int j = 0; j < 8; ++j) {                 // C/D: row = j + 8*half
        float sv = cs[j];
        int row = wave * 16 + j + 8 * half;
        attnH[(size_t)(r0 + row) * SDIM + col] = sv;   // raw scores (scratch)
        ssq[j]  += sv * sv;
        smax[j]  = fmaxf(smax[j], sv);
      }
    }
  }

  // reduce across the 16 lanes of each half (they share a row per C/D element)
#pragma unroll
  for (int m = 1; m <= 8; m <<= 1) {
#pragma unroll
    for (int j = 0; j < 8; ++j) {
      ssq[j] += __shfl_xor(ssq[j], m, 32);
      smax[j] = fmaxf(smax[j], __shfl_xor(smax[j], m, 32));
    }
  }
  if (ln == 0) {
#pragma unroll
    for (int j = 0; j < 8; ++j) {
      int row = wave * 16 + j + 8 * half;
      rowSumSq[row] = ssq[j];
      rowMax[row]   = smax[j];
    }
  }
  __syncthreads();

  // ---------------- Pass B: exp + row sums + PV via WMMA ----------------------
  const int myRow = wave * 16 + ln;                 // A-frag row = lane%16
  const float scale = 1.0f / fmaxf(sqrtf(rowSumSq[myRow]), 1e-20f);
  const float mterm = rowMax[myRow] * scale;
  float rsum = 0.0f;
  v8f oa[4];
#pragma unroll
  for (int s = 0; s < 4; ++s) oa[s] = (v8f){0.f,0.f,0.f,0.f,0.f,0.f,0.f,0.f};
  float* attnRow = attnH + (size_t)(r0 + myRow) * SDIM;

  for (int kt = 0; kt < SDIM / 32; ++kt) {
    __syncthreads();
    { // stage V tile (32 rows) transposed [d][t] as bf16; ds_store_b64
      const float* src = vh + (size_t)kt * 32 * DDIM;
#pragma unroll
      for (int i = 0; i < 4; ++i) {
        int idx = i * 128 + tid;                    // 512 tasks, d coalesced
        int d = idx & 63, t4 = idx >> 6;            // t = 4*t4 .. 4*t4+3
        float f0 = src[(4 * t4 + 0) * DDIM + d];
        float f1 = src[(4 * t4 + 1) * DDIM + d];
        float f2 = src[(4 * t4 + 2) * DDIM + d];
        float f3 = src[(4 * t4 + 3) * DDIM + d];
        uint2 p = make_uint2(pack2bf(f0, f1), pack2bf(f2, f3));
        *(uint2*)(lVt + d * 32 + 4 * t4) = p;       // 8-byte aligned
      }
    }
    __syncthreads();

    // P fragment: read raw scores as two contiguous 8-float runs per lane,
    // exponentiate, store unnormalized exp back (float4 each way).
    BFrag pa;
    {
      int cbase = kt * 32 + 8 * half;               // 32B aligned
      float4 r0a = *(float4*)(attnRow + cbase + 0);
      float4 r0b = *(float4*)(attnRow + cbase + 4);
      float4 r1a = *(float4*)(attnRow + cbase + 16);
      float4 r1b = *(float4*)(attnRow + cbase + 20);
#define EXP4(r)                                                         \
      r.x = __expf(r.x * scale - mterm); r.y = __expf(r.y * scale - mterm); \
      r.z = __expf(r.z * scale - mterm); r.w = __expf(r.w * scale - mterm); \
      rsum += (r.x + r.y) + (r.z + r.w)
      EXP4(r0a); EXP4(r0b); EXP4(r1a); EXP4(r1b);
#undef EXP4
      *(float4*)(attnRow + cbase + 0)  = r0a;
      *(float4*)(attnRow + cbase + 4)  = r0b;
      *(float4*)(attnRow + cbase + 16) = r1a;
      *(float4*)(attnRow + cbase + 20) = r1b;
      pa.u[0] = pack2bf(r0a.x, r0a.y); pa.u[1] = pack2bf(r0a.z, r0a.w);
      pa.u[2] = pack2bf(r0b.x, r0b.y); pa.u[3] = pack2bf(r0b.z, r0b.w);
      pa.u[4] = pack2bf(r1a.x, r1a.y); pa.u[5] = pack2bf(r1a.z, r1a.w);
      pa.u[6] = pack2bf(r1b.x, r1b.y); pa.u[7] = pack2bf(r1b.z, r1b.w);
    }

    const uint4* vb4 = (const uint4*)lVt;
#pragma unroll
    for (int s = 0; s < 4; ++s) {                   // four 16-wide D subtiles
      BFrag bv;
      int d = s * 16 + ln;
      bv.q[0] = vb4[d * 4 + half];                  // ds_load_b128 x2
      bv.q[1] = vb4[d * 4 + 2 + half];
      oa[s] = __builtin_amdgcn_wmma_f32_16x16x32_bf16(false, pa.v, false, bv.v,
                                                      (short)0, oa[s], false, false);
    }
  }

  // row sums: lanes l and l+16 share a row in the A-frag layout
  rsum += __shfl_xor(rsum, 16, 32);
  if (lane < 16) rowSum[wave * 16 + lane] = rsum;
  __threadfence();
  __syncthreads();

  // ---------------- outputs: out = acc / rowsum -------------------------------
#pragma unroll
  for (int s = 0; s < 4; ++s) {
#pragma unroll
    for (int j = 0; j < 8; ++j) {
      int row = wave * 16 + j + 8 * half;
      outH[(size_t)(r0 + row) * DDIM + s * 16 + ln] = oa[s][j] * (1.0f / rowSum[row]);
    }
  }

  // ---------------- Pass C: finalize attn = exp / rowsum ----------------------
  {
    int row = tid >> 1;
    int ch  = tid & 1;
    float inv = 1.0f / rowSum[row];
    float4* pr = (float4*)(attnH + (size_t)(r0 + row) * SDIM + ch * (SDIM / 2));
    for (int i = 0; i < SDIM / 8; ++i) {            // 512 float4 per thread
      float4 x = pr[i];
      x.x *= inv; x.y *= inv; x.z *= inv; x.w *= inv;
      pr[i] = x;
    }
  }
}

__global__ void copy_v_kernel(const float4* __restrict__ src,
                              float4* __restrict__ dst, int n4) {
  int i = blockIdx.x * blockDim.x + threadIdx.x;
  if (i < n4) dst[i] = src[i];
}

extern "C" void kernel_launch(void* const* d_in, const int* in_sizes, int n_in,
                              void* d_out, int out_size, void* d_ws, size_t ws_size,
                              hipStream_t stream) {
  const float* q = (const float*)d_in[0];
  const float* k = (const float*)d_in[1];
  const float* v = (const float*)d_in[2];
  float* out = (float*)d_out;

  dim3 grid(SDIM / BM, BHN);
  attn_fused<<<grid, NTHREADS, 0, stream>>>(q, k, v, out);

  const size_t vElems = (size_t)BHN * SDIM * DDIM;               // 4,194,304
  float* vdst = out + vElems + (size_t)BHN * SDIM * SDIM;        // third output
  int n4 = (int)(vElems / 4);
  copy_v_kernel<<<(n4 + 255) / 256, 256, 0, stream>>>((const float4*)v,
                                                      (float4*)vdst, n4);
}
